// GNLoss_46059229282600
// MI455X (gfx1250) — compile-verified, matrix operand order checked by reading, not run on hardware
//
#include <hip/hip_runtime.h>
#include <stdint.h>

// Problem constants (match reference)
#define B_      4
#define C_      128
#define N_      1024
#define M_      (B_ * N_)      // 4096 total match points
#define KD      128            // feature dim
#define MARGIN_ 1.0f
#define LAMDA_  0.3f
#define NTILES  (M_ / 16)      // 256 column tiles

typedef __attribute__((ext_vector_type(16))) __bf16       v16bf;
typedef __attribute__((ext_vector_type(8)))  float        v8f;
typedef __attribute__((ext_vector_type(4)))  unsigned int uint4v;

// ---------------------------------------------------------------- helpers

__device__ __forceinline__ unsigned short f32_to_bf16(float f) {
  union { float f; uint32_t u; } v; v.f = f;
  uint32_t u = v.u;
  u += 0x7FFFu + ((u >> 16) & 1u);   // round-to-nearest-even
  return (unsigned short)(u >> 16);
}

__device__ __forceinline__ float hash_unif(uint32_t seed, uint32_t idx) {
  uint32_t h = seed * 0x9E3779B9u + idx * 0x85EBCA6Bu + 0x7F4A7C15u;
  h ^= h >> 16; h *= 0x7FEB352Du;
  h ^= h >> 15; h *= 0x846CA68Bu;
  h ^= h >> 16;
  return ((float)(h >> 8)) * (2.0f / 16777216.0f) - 1.0f;   // [-1, 1)
}

__device__ __forceinline__ float bilin(const float* __restrict__ fm,
                                       int W, int H, float x, float y) {
  x = fminf(fmaxf(x, 0.0f), (float)(W - 1));
  y = fminf(fmaxf(y, 0.0f), (float)(H - 1));
  float fx0 = fmaxf(fminf(floorf(x), (float)(W - 2)), 0.0f);
  float fy0 = fmaxf(fminf(floorf(y), (float)(H - 2)), 0.0f);
  int x0 = (int)fx0, y0 = (int)fy0;
  float wx = x - fx0, wy = y - fy0;
  const float* p = fm + (size_t)y0 * W + x0;
  float f00 = p[0], f01 = p[1], f10 = p[W], f11 = p[W + 1];
  return f00 * (1.f - wx) * (1.f - wy) + f01 * wx * (1.f - wy)
       + f10 * (1.f - wx) * wy         + f11 * wx * wy;
}

// jnp.gradient semantics: one-sided at edges, central in interior.
__device__ __forceinline__ float gradx(const float* __restrict__ fm,
                                       int W, int y, int x) {
  const float* r = fm + (size_t)y * W;
  if (x == 0)     return r[1] - r[0];
  if (x == W - 1) return r[W - 1] - r[W - 2];
  return 0.5f * (r[x + 1] - r[x - 1]);
}
__device__ __forceinline__ float grady(const float* __restrict__ fm,
                                       int W, int H, int y, int x) {
  if (y == 0)     return fm[W + x] - fm[x];
  if (y == H - 1) return fm[(size_t)(H - 1) * W + x] - fm[(size_t)(H - 2) * W + x];
  return 0.5f * (fm[(size_t)(y + 1) * W + x] - fm[(size_t)(y - 1) * W + x]);
}

// Fused sample of f, d f/dx, d f/dy at one point — avoids materializing
// jnp.gradient() maps (saves ~0.7 GB of HBM traffic at level 3).
__device__ __forceinline__ void sample_fgxgy(const float* __restrict__ fm,
                                             int W, int H, float x, float y,
                                             float& f, float& gx, float& gy) {
  x = fminf(fmaxf(x, 0.0f), (float)(W - 1));
  y = fminf(fmaxf(y, 0.0f), (float)(H - 1));
  float fx0 = fmaxf(fminf(floorf(x), (float)(W - 2)), 0.0f);
  float fy0 = fmaxf(fminf(floorf(y), (float)(H - 2)), 0.0f);
  int x0 = (int)fx0, y0 = (int)fy0;
  float wx = x - fx0, wy = y - fy0;
  float w00 = (1.f - wx) * (1.f - wy), w01 = wx * (1.f - wy);
  float w10 = (1.f - wx) * wy,         w11 = wx * wy;
  const float* p = fm + (size_t)y0 * W + x0;
  f  = w00 * p[0] + w01 * p[1] + w10 * p[W] + w11 * p[W + 1];
  gx = w00 * gradx(fm, W, y0,     x0) + w01 * gradx(fm, W, y0,     x0 + 1)
     + w10 * gradx(fm, W, y0 + 1, x0) + w11 * gradx(fm, W, y0 + 1, x0 + 1);
  gy = w00 * grady(fm, W, H, y0,     x0) + w01 * grady(fm, W, H, y0,     x0 + 1)
     + w10 * grady(fm, W, H, y0 + 1, x0) + w11 * grady(fm, W, H, y0 + 1, x0 + 1);
}

// Load one v16bf WMMA operand half for a given lane:
// two contiguous 8x bf16 (16 B) chunks at element offsets kA and kB.
__device__ __forceinline__ v16bf load_k16(const unsigned short* __restrict__ row,
                                          int kA, int kB) {
  union { v16bf v; uint4v q[2]; } u;
  u.q[0] = *reinterpret_cast<const uint4v*>(row + kA);
  u.q[1] = *reinterpret_cast<const uint4v*>(row + kB);
  return u.v;
}

// ---------------------------------------------------------------- kernels

// Seed the scalar loss with the closed-form constant:
// sum over 4 levels of LAMDA * (2/7) * B*N * log(2*pi)
__global__ void gn_init_out(float* __restrict__ out) {
  out[0] = 4.0f * LAMDA_ * (2.0f / 7.0f) * (float)(B_ * N_) * 1.8378770664093453f;
}

// Bilinear-sample the level-3 features at matches/2, emit bf16 rows for the
// WMMA GEMM; optionally emit per-point squared norms (B side only).
__global__ __launch_bounds__(128)
void gn_sample_bf16(const float* __restrict__ F, const float* __restrict__ pts,
                    unsigned short* __restrict__ outBf,
                    float* __restrict__ norms, int computeNorm) {
  __shared__ float red[C_];
  const int m = blockIdx.x;       // point 0..4095
  const int c = threadIdx.x;      // channel 0..127
  const int b = m >> 10;          // batch = m / N
  float x = pts[2 * m]     * 0.5f;
  float y = pts[2 * m + 1] * 0.5f;
  const float* fm = F + (size_t)(b * C_ + c) * (256 * 256);
  float v = bilin(fm, 256, 256, x, y);
  outBf[(size_t)m * KD + c] = f32_to_bf16(v);
  if (computeNorm) {
    red[c] = v * v;
    __syncthreads();
    for (int s = 64; s > 0; s >>= 1) {
      if (c < s) red[c] += red[c + s];
      __syncthreads();
    }
    if (c == 0) norms[m] = red[0];
  }
}

// Nearest-neighbor argmin via bf16 WMMA. One wave owns a 16-row stripe and
// streams all 256 column tiles, software-pipelined with explicit double
// buffering (x/y tile buffers both live) so B-tile loads overlap the matrix
// pipe instead of serializing on s_wait_loadcnt 0.
// Hot loop keeps a PRIVATE per-lane running argmin over the lane's column
// residue (col ≡ lane mod 16); the 16-lane butterfly merge runs once at end.
__global__ __launch_bounds__(128)
void gn_nn_argmin_wmma(const unsigned short* __restrict__ Abf,
                       const unsigned short* __restrict__ Bbf,
                       const float* __restrict__ nb,
                       int* __restrict__ jidx) {
  const int lane    = threadIdx.x & 31;
  const int waveIdx = threadIdx.x >> 5;
  const int stripe  = blockIdx.x * 4 + waveIdx;
  const int rowbase = stripe * 16;
  const int half    = lane >> 4;      // K-half per 16-bit A/B VGPR layout
  const int l       = lane & 15;      // row (A) / column (B,C) within tile

  const int k0a =   0 + 8 * half, k0b =  16 + 8 * half;
  const int k1a =  32 + 8 * half, k1b =  48 + 8 * half;
  const int k2a =  64 + 8 * half, k2b =  80 + 8 * half;
  const int k3a =  96 + 8 * half, k3b = 112 + 8 * half;

  // A operand: row (rowbase + l); lane<16 holds K {0-7,16-23}+32s,
  // lane>=16 holds K {8-15,24-31}+32s — matches 16-bit A-matrix layout.
  const unsigned short* arow = Abf + (size_t)(rowbase + l) * KD;
  const v16bf a0 = load_k16(arow, k0a, k0b);
  const v16bf a1 = load_k16(arow, k1a, k1b);
  const v16bf a2 = load_k16(arow, k2a, k2b);
  const v16bf a3 = load_k16(arow, k3a, k3b);

  float minv[8];
  int   mini[8];
#pragma unroll
  for (int r = 0; r < 8; ++r) { minv[r] = 3.402823466e38f; mini[r] = 0; }

  // Prologue: buffer X <- tile 0.
  const unsigned short* brow0 = Bbf + (size_t)l * KD;
  v16bf x0 = load_k16(brow0, k0a, k0b);
  v16bf x1 = load_k16(brow0, k1a, k1b);
  v16bf x2 = load_k16(brow0, k2a, k2b);
  v16bf x3 = load_k16(brow0, k3a, k3b);

  for (int ct = 0; ct < NTILES; ct += 2) {
    // Issue loads for tile ct+1 into buffer Y (independent of X-WMMAs).
    const unsigned short* browY = Bbf + (size_t)((ct + 1) * 16 + l) * KD;
    const v16bf y0 = load_k16(browY, k0a, k0b);
    const v16bf y1 = load_k16(browY, k1a, k1b);
    const v16bf y2 = load_k16(browY, k2a, k2b);
    const v16bf y3 = load_k16(browY, k3a, k3b);
    if (ct + 2 < NTILES)
      __builtin_prefetch(Bbf + (size_t)((ct + 2) * 16 + l) * KD, 0, 1);

    // Compute tile ct from X while Y loads are in flight.
    v8f c0 = {};
    c0 = __builtin_amdgcn_wmma_f32_16x16x32_bf16(false, a0, false, x0, (short)0, c0, false, false);
    c0 = __builtin_amdgcn_wmma_f32_16x16x32_bf16(false, a1, false, x1, (short)0, c0, false, false);
    c0 = __builtin_amdgcn_wmma_f32_16x16x32_bf16(false, a2, false, x2, (short)0, c0, false, false);
    c0 = __builtin_amdgcn_wmma_f32_16x16x32_bf16(false, a3, false, x3, (short)0, c0, false, false);
    {
      // C layout: VGPR r = row r (lanes 0-15) / row r+8 (lanes 16-31),
      // lane&15 = column. Per-lane running argmin; no cross-lane ops.
      const float colNorm = nb[ct * 16 + l];
      const int   jcol    = ct * 16 + l;
#pragma unroll
      for (int r = 0; r < 8; ++r) {
        const float v = fmaf(-2.0f, c0[r], colNorm);
        if (v < minv[r]) { minv[r] = v; mini[r] = jcol; }
      }
    }

    // Refill X with tile ct+2 (safe tail reload), independent of Y-WMMAs.
    const int ctX = (ct + 2 < NTILES) ? (ct + 2) : ct;
    const unsigned short* browX = Bbf + (size_t)(ctX * 16 + l) * KD;
    x0 = load_k16(browX, k0a, k0b);
    x1 = load_k16(browX, k1a, k1b);
    x2 = load_k16(browX, k2a, k2b);
    x3 = load_k16(browX, k3a, k3b);

    // Compute tile ct+1 from Y while X refill is in flight.
    v8f c1 = {};
    c1 = __builtin_amdgcn_wmma_f32_16x16x32_bf16(false, a0, false, y0, (short)0, c1, false, false);
    c1 = __builtin_amdgcn_wmma_f32_16x16x32_bf16(false, a1, false, y1, (short)0, c1, false, false);
    c1 = __builtin_amdgcn_wmma_f32_16x16x32_bf16(false, a2, false, y2, (short)0, c1, false, false);
    c1 = __builtin_amdgcn_wmma_f32_16x16x32_bf16(false, a3, false, y3, (short)0, c1, false, false);
    {
      const float colNorm = nb[(ct + 1) * 16 + l];
      const int   jcol    = (ct + 1) * 16 + l;
#pragma unroll
      for (int r = 0; r < 8; ++r) {
        const float v = fmaf(-2.0f, c1[r], colNorm);
        if (v < minv[r]) { minv[r] = v; mini[r] = jcol; }
      }
    }
  }

  // One-time butterfly merge of the 16 per-lane column subsets.
#pragma unroll
  for (int r = 0; r < 8; ++r) {
    float v = minv[r];
    int   j = mini[r];
#pragma unroll
    for (int s = 8; s >= 1; s >>= 1) {
      const float ov = __shfl_xor(v, s, 32);
      const int   oj = __shfl_xor(j, s, 32);
      if (ov < v) { v = ov; j = oj; }
    }
    if (l == 0) jidx[rowbase + half * 8 + r] = j;
  }
}

// Fused per-level loss: contrastive pos/neg + Gauss-Newton refinement terms.
// One wave per match point; lane covers 4 channels.
__global__ __launch_bounds__(256)
void gn_level_loss(const float* __restrict__ Fa, const float* __restrict__ Fb,
                   const float* __restrict__ ma, const float* __restrict__ mb,
                   const int* __restrict__ jidx, float* __restrict__ out,
                   int Wl, float invScale, int level) {
  const int gtid = blockIdx.x * blockDim.x + threadIdx.x;
  const int m    = gtid >> 5;
  const int lane = threadIdx.x & 31;
  if (m >= M_) return;
  const int Hl    = Wl;
  const int plane = Wl * Hl;
  const int b     = m >> 10;

  const float pax = ma[2 * m]     * invScale, pay = ma[2 * m + 1] * invScale;
  const float pbx = mb[2 * m]     * invScale, pby = mb[2 * m + 1] * invScale;
  const int   j   = jidx[m];
  const float nbx = mb[2 * j]     * invScale, nby = mb[2 * j + 1] * invScale;
  // xs = ub + 3*U(-1,1)  (hash-based deterministic surrogate for jax PRNG)
  const float px  = 3.0f * hash_unif(0x9E37u + (uint32_t)level, 2u * m);
  const float py  = 3.0f * hash_unif(0x9E37u + (uint32_t)level, 2u * m + 1u);
  const float xsx = pbx + px, xsy = pby + py;

  float lp = 0.f, ln = 0.f;
  float Jxx = 0.f, Jxy = 0.f, Jyy = 0.f, bx = 0.f, by = 0.f;

#pragma unroll
  for (int k = 0; k < 4; ++k) {
    const int c = lane + 32 * k;
    const float* fa_c = Fa + (size_t)(b * C_ + c) * plane;
    const float* fb_c = Fb + (size_t)(b * C_ + c) * plane;

    const float fap = bilin(fa_c, Wl, Hl, pax, pay);   // pos A (== neg A coords)
    const float fbp = bilin(fb_c, Wl, Hl, pbx, pby);   // pos B
    const float fbn = bilin(fb_c, Wl, Hl, nbx, nby);   // neg B (NN-permuted)

    const float dp = fap - fbp;            lp += dp * dp;
    const float dn = fap - fbn;
    const float mg = fmaxf(MARGIN_ - dn, 0.0f);
    ln += mg * mg;

    float fs, gx, gy;
    sample_fgxgy(fb_c, Wl, Hl, xsx, xsy, fs, gx, gy);
    const float r = fs - fap;
    Jxx += gx * gx; Jxy += gx * gy; Jyy += gy * gy;
    bx  += gx * r;  by  += gy * r;
  }

  // wave32 butterfly reduction over channels
  for (int s = 16; s >= 1; s >>= 1) {
    lp  += __shfl_xor(lp,  s, 32);
    ln  += __shfl_xor(ln,  s, 32);
    Jxx += __shfl_xor(Jxx, s, 32);
    Jxy += __shfl_xor(Jxy, s, 32);
    Jyy += __shfl_xor(Jyy, s, 32);
    bx  += __shfl_xor(bx,  s, 32);
    by  += __shfl_xor(by,  s, 32);
  }

  if (lane == 0) {
    const float Hxx = Jxx + 1e-9f, Hyy = Jyy + 1e-9f, Hxy = Jxy;
    const float det = Hxx * Hyy - Hxy * Hxy;
    const float inv = 1.0f / det;
    // s = H^-1 b ; miu = xs - s ; d = ub - miu = s - (xs - ub) = s - p
    const float sx = (Hyy * bx - Hxy * by) * inv;
    const float sy = (Hxx * by - Hxy * bx) * inv;
    const float dx = sx - px, dy = sy - py;
    const float e1 = 0.5f * (Hxx * dx * dx + 2.0f * Hxy * dx * dy + Hyy * dy * dy);
    const float lg = e1 - (1.0f / 7.0f) * logf(det);   // + const handled in init
    const float contrib = (lp + ln) * (1.0f / (float)N_) + LAMDA_ * lg;
    atomicAdd(out, contrib);
  }
}

// ---------------------------------------------------------------- launch

extern "C" void kernel_launch(void* const* d_in, const int* in_sizes, int n_in,
                              void* d_out, int out_size, void* d_ws, size_t ws_size,
                              hipStream_t stream) {
  (void)in_sizes; (void)n_in; (void)out_size; (void)ws_size;
  // setup_inputs() dict order: Fa0,Fb0,Fa1,Fb1,Fa2,Fb2,Fa3,Fb3,matches_a,matches_b
  const float* Fa[4] = { (const float*)d_in[0], (const float*)d_in[2],
                         (const float*)d_in[4], (const float*)d_in[6] };
  const float* Fb[4] = { (const float*)d_in[1], (const float*)d_in[3],
                         (const float*)d_in[5], (const float*)d_in[7] };
  const float* ma = (const float*)d_in[8];
  const float* mb = (const float*)d_in[9];
  float* out = (float*)d_out;

  // Workspace layout (~2.13 MB): A bf16 | B bf16 | |B|^2 norms | argmin idx
  char* ws = (char*)d_ws;
  unsigned short* Abf = (unsigned short*)(ws);
  unsigned short* Bbf = (unsigned short*)(ws + (size_t)M_ * KD * 2);
  float*          nb  = (float*)(ws + (size_t)2 * M_ * KD * 2);
  int*            jix = (int*)  (ws + (size_t)2 * M_ * KD * 2 + (size_t)M_ * 4);

  gn_init_out<<<1, 1, 0, stream>>>(out);

  gn_sample_bf16<<<M_, C_, 0, stream>>>(Fa[3], ma, Abf, nullptr, 0);
  gn_sample_bf16<<<M_, C_, 0, stream>>>(Fb[3], mb, Bbf, nb, 1);

  // 256 stripes of 16 rows, 4 waves (128 threads) per block -> 64 blocks
  gn_nn_argmin_wmma<<<M_ / 16 / 4, 128, 0, stream>>>(Abf, Bbf, nb, jix);

  for (int i = 0; i < 4; ++i) {
    const int   s        = 1 << (3 - i);        // level = 2^(3-i)
    const int   Wl       = 256 >> (3 - i);
    const float invScale = 1.0f / (2.0f * (float)s);
    gn_level_loss<<<(M_ * 32) / 256, 256, 0, stream>>>(
        Fa[i], Fb[i], ma, mb, jix, out, Wl, invScale, i);
  }
}
